// GuidedDiffusionRGC_69578470195202
// MI455X (gfx1250) — compile-verified, matrix operand order checked by reading, not run on hardware
//
#include <hip/hip_runtime.h>

#define NN   32768          // num_nodes = B*N
#define EDG  524288         // edges
#define RR   8              // relations
#define NB   4              // bases
#define DD   256            // D == HIDDEN
#define RNN  (RR * NN)      // segments
#define SLOTS (RR + 1)      // 8 relations + root

typedef __attribute__((ext_vector_type(16))) __bf16 v16bf;
typedef __attribute__((ext_vector_type(8)))  __bf16 v8bf;
typedef __attribute__((ext_vector_type(4)))  __bf16 v4bf;
typedef __attribute__((ext_vector_type(8)))  float  v8f;

// ---------------------------------------------------------------------------
// W_r = sum_b comp[r,b] * basis[b]  (r<8), slot 8 = root.  Stored TRANSPOSED
// as bf16 [rel][out][in] so GEMM B-tiles are contiguous along K(=in).
// ---------------------------------------------------------------------------
__global__ void k_combine(const float* __restrict__ basis,
                          const float* __restrict__ comp,
                          const float* __restrict__ root,
                          __bf16* __restrict__ Wt)
{
    int idx = blockIdx.x * 256 + threadIdx.x;   // 9*256*256 total
    int r   = idx >> 16;                        // relation slot (0..8)
    int rem = idx & 65535;
    int o   = rem >> 8;                         // output dim
    int i   = rem & 255;                        // input dim (K)
    float w;
    if (r < RR) {
        w = 0.0f;
        #pragma unroll
        for (int b = 0; b < NB; ++b)
            w += comp[r * NB + b] * basis[((b << 8) + i) * DD + o];
    } else {
        w = root[i * DD + o];
    }
    Wt[idx] = (__bf16)w;                        // idx == (r*256 + o)*256 + i
}

// ---------------------------------------------------------------------------
// Per-(relation,dst) edge counts -> reciprocals; root slot (seg >= RNN) = 1.
// ---------------------------------------------------------------------------
__global__ void k_count(const int* __restrict__ dst, const int* __restrict__ rel,
                        float* __restrict__ cnt)
{
    int e = blockIdx.x * 256 + threadIdx.x;
    if (e < EDG)
        atomicAdd(&cnt[(long)rel[e] * NN + dst[e]], 1.0f);
}

__global__ void k_inv(float* __restrict__ cnt)
{
    int i = blockIdx.x * 256 + threadIdx.x;     // SLOTS*NN threads
    if (i < RNN)
        cnt[i] = 1.0f / fmaxf(cnt[i], 1.0f);
    else if (i < SLOTS * NN)
        cnt[i] = 1.0f;                          // root path: scale 1
}

// ---------------------------------------------------------------------------
// Scatter-add raw source features into per-(rel,dst) segment sums.
// 64 threads per edge, float4 per thread, global_atomic_add_f32.
// ---------------------------------------------------------------------------
__global__ void k_scatter(const float* __restrict__ h,
                          const int* __restrict__ src,
                          const int* __restrict__ dst,
                          const int* __restrict__ rel,
                          float* __restrict__ S)
{
    long gid = (long)blockIdx.x * 256 + threadIdx.x;   // E*64 threads
    int  e   = (int)(gid >> 6);
    int  q   = (int)(gid & 63);
    int  s   = src[e];
    long seg = (long)rel[e] * NN + dst[e];
    float4 v = *(const float4*)(h + (long)s * DD + q * 4);
    float* p = S + seg * DD + q * 4;
    atomicAdd(p + 0, v.x);
    atomicAdd(p + 1, v.y);
    atomicAdd(p + 2, v.z);
    atomicAdd(p + 3, v.w);
}

// ---------------------------------------------------------------------------
// Fused GEMM:  out[n,o] = act( bias[o] + sum_{r<8} (S_r[n,:]*inv_r[n]) @ W_r
//                                       + h[n,:] @ root )
// K = 9*256 = 2304, bf16 WMMA 16x16x32, f32 accumulate.
// Double-buffered LDS (ping-pong), global loads for step k+1 overlap the
// WMMAs of step k; one barrier per K-step.  Branch-free A fill: base pointer
// is a uniform scalar select (for layer 1, h sits contiguously after S so
// both arms coincide); mean scale comes from inv[seg] (root slot == 1.0).
// ---------------------------------------------------------------------------
__global__ __launch_bounds__(256)
void k_gemm(const float* __restrict__ S, const float* __restrict__ invc,
            const float* __restrict__ h, const __bf16* __restrict__ Wt,
            const float* __restrict__ bias, float* __restrict__ out,
            int do_relu)
{
    __shared__ __align__(16) __bf16 As[2][128 * 32];   // (row, k)  2 x 8 KB
    __shared__ __align__(16) __bf16 Bs[2][128 * 32];   // (col, k)  2 x 8 KB

    const int t       = threadIdx.x;
    const int rowBase = blockIdx.x * 128;
    const int colBase = blockIdx.y * 128;
    const int lane    = t & 31;
    const int wave    = t >> 5;
    const int half    = lane >> 4;     // 0: K 0-7/16-23, 1: K 8-15/24-31
    const int lr      = lane & 15;

    float4 av[4]; float asc[4]; uint4 bv[2];

    // ---- stage K-step `ks` from global into registers ----
    auto load_tiles = [&](int ks) {
        const int rel = ks >> 3;
        const int k0  = (ks & 7) << 5;
        // uniform scalar select; for layer 1 both arms are the same address
        const float* base = (rel < RR) ? S : (h - (size_t)RR * NN * DD);
        #pragma unroll
        for (int it = 0; it < 4; ++it) {
            int idx = t + it * 256;            // 1024 float4 slots
            int rl  = idx >> 3;                // 0..127
            int kq  = idx & 7;                 // 0..7
            size_t seg = (size_t)rel * NN + rowBase + rl;
            av[it]  = *(const float4*)(base + seg * DD + k0 + kq * 4);
            asc[it] = invc[seg];
        }
        #pragma unroll
        for (int it = 0; it < 2; ++it) {
            int idx = t + it * 256;            // 512 x 16B slots
            int ol  = idx >> 2;                // 0..127
            int ch  = idx & 3;                 // 0..3 (8 bf16 each)
            bv[it] = *(const uint4*)
                (Wt + ((size_t)(rel * 256 + colBase + ol)) * DD + k0 + ch * 8);
        }
    };
    // ---- convert + store staged registers into LDS buffer `buf` ----
    auto store_tiles = [&](int buf) {
        #pragma unroll
        for (int it = 0; it < 4; ++it) {
            int idx = t + it * 256;
            int rl  = idx >> 3;
            int kq  = idx & 7;
            v4bf p;
            p[0] = (__bf16)(av[it].x * asc[it]);
            p[1] = (__bf16)(av[it].y * asc[it]);
            p[2] = (__bf16)(av[it].z * asc[it]);
            p[3] = (__bf16)(av[it].w * asc[it]);
            *(v4bf*)&As[buf][rl * 32 + kq * 4] = p;
        }
        #pragma unroll
        for (int it = 0; it < 2; ++it) {
            int idx = t + it * 256;
            int ol  = idx >> 2;
            int ch  = idx & 3;
            *(uint4*)&Bs[buf][ol * 32 + ch * 8] = bv[it];
        }
    };

    v8f acc[8] = {};

    load_tiles(0);
    store_tiles(0);
    __syncthreads();

    const int arow = wave * 16 + lr;
    const int kb   = half * 8;

    for (int ks = 0; ks < 72; ++ks) {          // 9 relation-slots * 8 K-chunks
        const int buf = ks & 1;
        if (ks < 71) load_tiles(ks + 1);       // HBM latency hides under WMMA

        // A fragment per ISA 16-bit layout
        v8bf alo = *(const v8bf*)&As[buf][arow * 32 + kb];
        v8bf ahi = *(const v8bf*)&As[buf][arow * 32 + kb + 16];
        v16bf a  = __builtin_shufflevector(alo, ahi,
                       0,1,2,3,4,5,6,7,8,9,10,11,12,13,14,15);
        // batch all B fragments, then burst the WMMAs
        v16bf bfr[8];
        #pragma unroll
        for (int j = 0; j < 8; ++j) {
            int brow = j * 16 + lr;
            v8bf blo = *(const v8bf*)&Bs[buf][brow * 32 + kb];
            v8bf bhi = *(const v8bf*)&Bs[buf][brow * 32 + kb + 16];
            bfr[j] = __builtin_shufflevector(blo, bhi,
                         0,1,2,3,4,5,6,7,8,9,10,11,12,13,14,15);
        }
        #pragma unroll
        for (int j = 0; j < 8; ++j)
            acc[j] = __builtin_amdgcn_wmma_f32_16x16x32_bf16(
                         false, a, false, bfr[j], (short)0, acc[j], false, false);

        if (ks < 71) {
            store_tiles(buf ^ 1);              // other buffer: no RAW on LDS
            __syncthreads();
        }
    }

    // ---- epilogue: C layout = VGPR g -> M = g + 8*half, N = lane&15 ----
    const int rbase = rowBase + wave * 16 + half * 8;
    #pragma unroll
    for (int j = 0; j < 8; ++j) {
        int col = colBase + j * 16 + lr;
        float bv2 = bias[col];
        #pragma unroll
        for (int g = 0; g < 8; ++g) {
            float r = acc[j][g] + bv2;
            if (do_relu) r = fmaxf(r, 0.0f);
            out[(long)(rbase + g) * DD + col] = r;
        }
    }
}

// ---------------------------------------------------------------------------
extern "C" void kernel_launch(void* const* d_in, const int* in_sizes, int n_in,
                              void* d_out, int out_size, void* d_ws, size_t ws_size,
                              hipStream_t stream)
{
    (void)in_sizes; (void)n_in; (void)out_size; (void)ws_size;

    const float* x      = (const float*)d_in[0];
    const int*   edge   = (const int*)d_in[1];     // [2, E]
    const int*   relc   = (const int*)d_in[2];
    const float* basis0 = (const float*)d_in[3];
    const float* comp0  = (const float*)d_in[4];
    const float* root0  = (const float*)d_in[5];
    const float* bias0  = (const float*)d_in[6];
    const float* basis1 = (const float*)d_in[7];
    const float* comp1  = (const float*)d_in[8];
    const float* root1  = (const float*)d_in[9];
    const float* bias1  = (const float*)d_in[10];
    const int* src = edge;
    const int* dst = edge + EDG;

    const size_t SB = (size_t)RNN * DD * sizeof(float);    // 256 MB segment sums
    const size_t HB = (size_t)NN * DD * sizeof(float);     //  32 MB h1 (MUST follow S)
    const size_t IB = (size_t)SLOTS * NN * sizeof(float);  // 1.125 MB inv counts

    char*  ws  = (char*)d_ws;
    float* S   = (float*)ws;                     // [RNN][DD]
    float* h1  = (float*)(ws + SB);              // contiguous 9th slot of S
    float* inv = (float*)(ws + SB + HB);         // [SLOTS*NN]
    __bf16* Wt = (__bf16*)(ws + SB + HB + IB);   // [9][256][256] bf16
    float* outp = (float*)d_out;

    // graph shared across layers: counts once
    hipMemsetAsync(S,   0, SB, stream);
    hipMemsetAsync(inv, 0, (size_t)RNN * sizeof(float), stream);
    k_count<<<EDG / 256, 256, 0, stream>>>(dst, relc, inv);
    k_inv<<<(SLOTS * NN) / 256, 256, 0, stream>>>(inv);

    // ---- layer 0: D -> HIDDEN, relu ----
    k_combine<<<9 * 256, 256, 0, stream>>>(basis0, comp0, root0, Wt);
    k_scatter<<<EDG / 4, 256, 0, stream>>>(x, src, dst, relc, S);   // E*64 thr
    k_gemm<<<dim3(NN / 128, 2), 256, 0, stream>>>(S, inv, x, Wt, bias0, h1, 1);

    // ---- layer 1: HIDDEN -> D ----
    hipMemsetAsync(S, 0, SB, stream);
    k_combine<<<9 * 256, 256, 0, stream>>>(basis1, comp1, root1, Wt);
    k_scatter<<<EDG / 4, 256, 0, stream>>>(h1, src, dst, relc, S);
    k_gemm<<<dim3(NN / 128, 2), 256, 0, stream>>>(S, inv, h1, Wt, bias1, outp, 0);
}
